// IHCModel_26422638805413
// MI455X (gfx1250) — matrix operand order repaired and checked by reading, not compiled, not run on hardware
//
#include <hip/hip_runtime.h>
#include <hip/hip_bf16.h>
#include <math.h>

// Problem constants (from reference): B=2, T=2048, C=1024, V=8192, L=2, H=16, d=64
#define BQ 2
#define TQ 2048
#define CQ 1024
#define VQ 8192
#define LQ 2
#define HQ 16
#define DQ 64
#define MQ (BQ * TQ)   // 4096 rows of activations

typedef __bf16 bf16_t;
typedef bf16_t v16bf __attribute__((ext_vector_type(16)));
typedef float  v8f   __attribute__((ext_vector_type(8)));

union Frag {
  v16bf v;
  uint4 q[2];
  unsigned short u[16];
};

__device__ __forceinline__ unsigned short f2bf(float f) {
  unsigned u = __float_as_uint(f);
  u += 0x7FFFu + ((u >> 16) & 1u);   // round-to-nearest-even
  return (unsigned short)(u >> 16);
}

// ---------------------------------------------------------------------------
// WMMA fragment loaders (bf16, 16x16x32).
//
// A-frag (16 rows x 32 K), source row-major with leading dim `ld`:
//   lane<16 : row = lane,     K = {k0..k0+7, k0+16..k0+23}
//   lane>=16: row = lane-16,  K = {k0+8..k0+15, k0+24..k0+31}
// (CDNA5 ISA 7.12.2, 16-bit A-matrix table)
__device__ __forceinline__ Frag load_a(const unsigned short* base, int ld, int k0) {
  int lane = threadIdx.x & 31;
  int r = lane & 15, half = lane >> 4;
  const unsigned short* p = base + (size_t)r * ld + k0 + half * 8;
  Frag f;
  f.q[0] = *reinterpret_cast<const uint4*>(p);        // K = +0..+7
  f.q[1] = *reinterpret_cast<const uint4*>(p + 16);   // K = +16..+23
  return f;
}

// B-frag (32 K x 16 cols) where memory holds the matrix as [N][K] row-major
// (i.e. B = W^T with W row-major), leading dim `ld`:
//   lane<16 : col n = lane,     K = {k0..k0+7, k0+16..k0+23}
//   lane>=16: col n = lane-16,  K = {k0+8..k0+15, k0+24..k0+31}
// (same K order as the A-frag)
__device__ __forceinline__ Frag load_b(const unsigned short* base, int ld, int k0) {
  int lane = threadIdx.x & 31;
  int r = lane & 15, half = lane >> 4;
  const unsigned short* p = base + (size_t)r * ld + k0 + half * 8;
  Frag f;
  f.q[0] = *reinterpret_cast<const uint4*>(p);        // K = +0..+7
  f.q[1] = *reinterpret_cast<const uint4*>(p + 16);   // K = +16..+23
  return f;
}

// ---------------------------------------------------------------------------
// Generic GEMM: Y[M,N] = X[M,K] * W[N,K]^T (+ bias[n]) (+ residual[m,n])
// X, W are bf16 (ushort bit pattern); accumulate fp32; optional f32/bf16 out.
// One wave -> 32x64 output tile: 2 A-frags x 4 B-frags -> 8 WMMAs / k-step,
// software-pipelined with a PEELED final iteration: the steady-state loop
// prefetches k0+32 unconditionally (plain pointer increments, no cselect
// address chains) while the current step's 8 WMMAs run on resident frags.
// Block = 256 threads = 8 waves stacked along M. grid = (N/64, M/256).
__global__ __launch_bounds__(256) void gemm_bf16_kernel(
    const unsigned short* __restrict__ X, const unsigned short* __restrict__ W,
    const float* __restrict__ bias, const float* __restrict__ residual,
    float* __restrict__ outF, unsigned short* __restrict__ outBf,
    int M, int N, int K) {
  int wave = threadIdx.x >> 5;
  int lane = threadIdx.x & 31;
  int r = lane & 15, half = lane >> 4;
  int row0 = (blockIdx.y * 8 + wave) * 32;
  int col0 = blockIdx.x * 64;
  if (row0 >= M || col0 >= N) return;

  const unsigned short* x0 = X + (size_t)row0 * K;
  const unsigned short* x1 = X + (size_t)(row0 + 16) * K;
  const unsigned short* w0 = W + (size_t)col0 * K;

  v8f acc[8] = {};
  Frag a0c = load_a(x0, K, 0);
  Frag a1c = load_a(x1, K, 0);
  Frag bc[4];
#pragma unroll
  for (int j = 0; j < 4; ++j) bc[j] = load_b(w0 + (size_t)(j * 16) * K, K, 0);

  int nIter = K >> 5;                       // K / 32 k-steps
  for (int it = 0; it < nIter - 1; ++it) {
    int kn = (it + 1) << 5;                 // unconditional next-step prefetch
    Frag a0n = load_a(x0, K, kn);
    Frag a1n = load_a(x1, K, kn);
    Frag bn[4];
#pragma unroll
    for (int j = 0; j < 4; ++j) bn[j] = load_b(w0 + (size_t)(j * 16) * K, K, kn);
    // 8 WMMAs on the already-resident fragments
#pragma unroll
    for (int j = 0; j < 4; ++j) {
      acc[j] = __builtin_amdgcn_wmma_f32_16x16x32_bf16(
          false, a0c.v, false, bc[j].v, (short)0, acc[j], false, false);
      acc[4 + j] = __builtin_amdgcn_wmma_f32_16x16x32_bf16(
          false, a1c.v, false, bc[j].v, (short)0, acc[4 + j], false, false);
    }
    a0c = a0n;
    a1c = a1n;
#pragma unroll
    for (int j = 0; j < 4; ++j) bc[j] = bn[j];
  }
  // peeled final k-step (no prefetch)
#pragma unroll
  for (int j = 0; j < 4; ++j) {
    acc[j] = __builtin_amdgcn_wmma_f32_16x16x32_bf16(
        false, a0c.v, false, bc[j].v, (short)0, acc[j], false, false);
    acc[4 + j] = __builtin_amdgcn_wmma_f32_16x16x32_bf16(
        false, a1c.v, false, bc[j].v, (short)0, acc[4 + j], false, false);
  }
  // C/D layout: VGPR p, lane<16 -> (m=p, n=lane); lane>=16 -> (m=p+8, n=lane-16)
#pragma unroll
  for (int hm = 0; hm < 2; ++hm) {
#pragma unroll
    for (int j = 0; j < 4; ++j) {
      int n = col0 + j * 16 + r;
      float bv = bias ? bias[n] : 0.f;
#pragma unroll
      for (int p = 0; p < 8; ++p) {
        int m = row0 + hm * 16 + half * 8 + p;
        size_t idx = (size_t)m * N + n;
        float val = acc[hm * 4 + j][p] + bv;
        if (residual) val += residual[idx];
        if (outF) outF[idx] = val;
        if (outBf) outBf[idx] = f2bf(val);
      }
    }
  }
}

// ---------------------------------------------------------------------------
// Flash attention, one wave = one (b, h, 16-query tile). Keys in blocks of 32.
// Q,K: bf16 [B,T,C] (head slice at h*D). Vt: bf16 [B,H,D,T]. ctx: bf16 [B,T,C].
__global__ __launch_bounds__(256) void attn_kernel(
    const unsigned short* __restrict__ Q, const unsigned short* __restrict__ Kmat,
    const unsigned short* __restrict__ Vt, unsigned short* __restrict__ ctx) {
  __shared__ __align__(16) unsigned short plds[8][16 * 32];  // per-wave P tile
  int wave = threadIdx.x >> 5;
  int lane = threadIdx.x & 31;
  int r = lane & 15, half = lane >> 4;

  int unit = blockIdx.x * 8 + wave;            // B*H*(T/16) units total
  int qt = unit & (TQ / 16 - 1);
  int h  = (unit / (TQ / 16)) & (HQ - 1);
  int b  = unit / ((TQ / 16) * HQ);
  int q0 = qt * 16;
  const float scale = 0.125f;                  // 1/sqrt(64)

  const unsigned short* qbase = Q + (size_t)(b * TQ + q0) * CQ + h * DQ;
  Frag aq0 = load_a(qbase, CQ, 0);             // Q dims 0..31
  Frag aq1 = load_a(qbase, CQ, 32);            // Q dims 32..63

  v8f o[4] = {};
  float mrow[8], lrow[8];
#pragma unroll
  for (int p = 0; p < 8; ++p) { mrow[p] = -INFINITY; lrow[p] = 0.f; }

  int nkb = (q0 + 15) / 32 + 1;                // causal: key blocks touching tile
  for (int kb = 0; kb < nkb; ++kb) {
    int kk = kb * 32;
    // S = Q * K^T for 32 keys (two 16-col tiles), K-dim = 64 -> 2 steps each
    v8f s[2] = {};
#pragma unroll
    for (int j = 0; j < 2; ++j) {
      const unsigned short* kbase = Kmat + (size_t)(b * TQ + kk + j * 16) * CQ + h * DQ;
      Frag b0 = load_b(kbase, CQ, 0);
      s[j] = __builtin_amdgcn_wmma_f32_16x16x32_bf16(
          false, aq0.v, false, b0.v, (short)0, s[j], false, false);
      Frag b1 = load_b(kbase, CQ, 32);
      s[j] = __builtin_amdgcn_wmma_f32_16x16x32_bf16(
          false, aq1.v, false, b1.v, (short)0, s[j], false, false);
    }
    // scale + causal mask + online softmax (row lives in one 16-lane half)
    float sv0[8], sv1[8], tmax[8];
#pragma unroll
    for (int p = 0; p < 8; ++p) {
      int m = q0 + half * 8 + p;
      float v0 = s[0][p] * scale;
      float v1 = s[1][p] * scale;
      if (kk + r > m)      v0 = -INFINITY;
      if (kk + 16 + r > m) v1 = -INFINITY;
      sv0[p] = v0; sv1[p] = v1;
      float t = fmaxf(v0, v1);
      t = fmaxf(t, __shfl_xor(t, 1, 32));
      t = fmaxf(t, __shfl_xor(t, 2, 32));
      t = fmaxf(t, __shfl_xor(t, 4, 32));
      t = fmaxf(t, __shfl_xor(t, 8, 32));
      tmax[p] = t;
    }
#pragma unroll
    for (int p = 0; p < 8; ++p) {
      float mn   = fmaxf(mrow[p], tmax[p]);
      float corr = __expf(mrow[p] - mn);
      mrow[p] = mn;
      float p0 = __expf(sv0[p] - mn);
      float p1 = __expf(sv1[p] - mn);
      float ts = p0 + p1;
      ts += __shfl_xor(ts, 1, 32);
      ts += __shfl_xor(ts, 2, 32);
      ts += __shfl_xor(ts, 4, 32);
      ts += __shfl_xor(ts, 8, 32);
      lrow[p] = lrow[p] * corr + ts;
#pragma unroll
      for (int j2 = 0; j2 < 4; ++j2) o[j2][p] *= corr;
      // stage P tile (16q x 32k) row-major bf16 in LDS for A-frag re-read
      plds[wave][(half * 8 + p) * 32 + r]      = f2bf(p0);
      plds[wave][(half * 8 + p) * 32 + 16 + r] = f2bf(p1);
    }
    asm volatile("s_wait_dscnt 0" ::: "memory");
    Frag pa = load_a(&plds[wave][0], 32, 0);
    // O += P * V  (V^T is [B,H,D,T] so B-frag loads are contiguous along T)
#pragma unroll
    for (int j2 = 0; j2 < 4; ++j2) {
      const unsigned short* vbase =
          Vt + (size_t)((b * HQ + h) * DQ + j2 * 16) * TQ + kk;
      Frag vb = load_b(vbase, TQ, 0);
      o[j2] = __builtin_amdgcn_wmma_f32_16x16x32_bf16(
          false, pa.v, false, vb.v, (short)0, o[j2], false, false);
    }
  }
  // normalize and store ctx (bf16)
#pragma unroll
  for (int j2 = 0; j2 < 4; ++j2) {
#pragma unroll
    for (int p = 0; p < 8; ++p) {
      int m = q0 + half * 8 + p;
      float val = o[j2][p] / lrow[p];
      ctx[(size_t)(b * TQ + m) * CQ + h * DQ + j2 * 16 + r] = f2bf(val);
    }
  }
}

// ---------------------------------------------------------------------------
__global__ void convert_f32_bf16_kernel(const float* __restrict__ src,
                                        unsigned short* __restrict__ dst, int n) {
  int i = blockIdx.x * 256 + threadIdx.x;
  if (i < n) dst[i] = f2bf(src[i]);
}

__global__ void embed_kernel(const int* __restrict__ x, const float* __restrict__ embed_w,
                             const float* __restrict__ pos,
                             float* __restrict__ hF, unsigned short* __restrict__ hBf) {
  int i = blockIdx.x * 256 + threadIdx.x;
  if (i >= MQ * CQ) return;
  int c = i & (CQ - 1);
  int bt = i / CQ;
  int t = bt & (TQ - 1);
  float v = embed_w[(size_t)x[bt] * CQ + c] + pos[(size_t)t * CQ + c];
  hF[i] = v;
  hBf[i] = f2bf(v);
}

// Vt[b,h,dim,t] = V[b,t,h*D+dim]   (bf16 -> bf16 gather-transpose)
__global__ void vtrans_kernel(const unsigned short* __restrict__ v,
                              unsigned short* __restrict__ vt) {
  int i = blockIdx.x * 256 + threadIdx.x;
  if (i >= BQ * HQ * DQ * TQ) return;
  int t = i & (TQ - 1);
  int rem = i / TQ;
  int dim = rem & (DQ - 1);
  rem /= DQ;
  int h = rem & (HQ - 1);
  int b = rem / HQ;
  vt[i] = v[(size_t)(b * TQ + t) * CQ + h * DQ + dim];
}

__global__ void zero_loss_kernel(float* loss) { *loss = 0.f; }

// One block per (b,t) row: lse over V=8192 logits, nll, atomic mean-accumulate.
__global__ __launch_bounds__(256) void loss_kernel(const float* __restrict__ logits,
                                                   const int* __restrict__ target,
                                                   float* __restrict__ loss) {
  __shared__ float red[256];
  int tid = threadIdx.x;
  const float* row = logits + (size_t)blockIdx.x * VQ;
  float mx = -INFINITY;
  for (int i = tid; i < VQ; i += 256) mx = fmaxf(mx, row[i]);
  red[tid] = mx;
  __syncthreads();
  for (int s = 128; s > 0; s >>= 1) {
    if (tid < s) red[tid] = fmaxf(red[tid], red[tid + s]);
    __syncthreads();
  }
  float m = red[0];
  __syncthreads();
  float sum = 0.f;
  for (int i = tid; i < VQ; i += 256) sum += __expf(row[i] - m);
  red[tid] = sum;
  __syncthreads();
  for (int s = 128; s > 0; s >>= 1) {
    if (tid < s) red[tid] += red[tid + s];
    __syncthreads();
  }
  if (tid == 0) {
    float lse = m + __logf(red[0]);
    float nll = lse - row[target[blockIdx.x]];
    atomicAdd(loss, nll * (1.0f / (float)MQ));
  }
}

// ---------------------------------------------------------------------------
extern "C" void kernel_launch(void* const* d_in, const int* in_sizes, int n_in,
                              void* d_out, int out_size, void* d_ws, size_t ws_size,
                              hipStream_t stream) {
  (void)in_sizes; (void)n_in; (void)out_size; (void)ws_size;
  const int*   x      = (const int*)d_in[0];
  const int*   target = (const int*)d_in[1];
  const float* embedw = (const float*)d_in[2];
  const float* pos    = (const float*)d_in[3];
  const float* Wq     = (const float*)d_in[4];
  const float* bq     = (const float*)d_in[5];
  const float* Wk     = (const float*)d_in[6];
  const float* bk     = (const float*)d_in[7];
  const float* Wv     = (const float*)d_in[8];
  const float* bv     = (const float*)d_in[9];
  const float* Wo     = (const float*)d_in[10];
  const float* bo     = (const float*)d_in[11];
  const float* Wu     = (const float*)d_in[12];

  float* logits = (float*)d_out;                      // [B,T,V]
  float* lossp  = (float*)d_out + (size_t)MQ * VQ;    // scalar

  // workspace carve-up (256B aligned)
  size_t off = 0;
  auto carve = [&](size_t bytes) -> void* {
    void* p = (char*)d_ws + off;
    off += (bytes + 255) & ~(size_t)255;
    return p;
  };
  unsigned short* wq_bf  = (unsigned short*)carve((size_t)LQ * CQ * CQ * 2);
  unsigned short* wk_bf  = (unsigned short*)carve((size_t)LQ * CQ * CQ * 2);
  unsigned short* wv_bf  = (unsigned short*)carve((size_t)LQ * CQ * CQ * 2);
  unsigned short* wo_bf  = (unsigned short*)carve((size_t)LQ * CQ * CQ * 2);
  unsigned short* wu_bf  = (unsigned short*)carve((size_t)VQ * CQ * 2);
  float*          h_f    = (float*)carve((size_t)MQ * CQ * 4);
  unsigned short* h_bf   = (unsigned short*)carve((size_t)MQ * CQ * 2);
  unsigned short* q_bf   = (unsigned short*)carve((size_t)MQ * CQ * 2);
  unsigned short* k_bf   = (unsigned short*)carve((size_t)MQ * CQ * 2);
  unsigned short* v_bf   = (unsigned short*)carve((size_t)MQ * CQ * 2);
  unsigned short* vt_bf  = (unsigned short*)carve((size_t)BQ * HQ * DQ * TQ * 2);
  unsigned short* ctx_bf = (unsigned short*)carve((size_t)MQ * CQ * 2);

  // 1) weights -> bf16
  {
    int n = LQ * CQ * CQ;
    convert_f32_bf16_kernel<<<(n + 255) / 256, 256, 0, stream>>>(Wq, wq_bf, n);
    convert_f32_bf16_kernel<<<(n + 255) / 256, 256, 0, stream>>>(Wk, wk_bf, n);
    convert_f32_bf16_kernel<<<(n + 255) / 256, 256, 0, stream>>>(Wv, wv_bf, n);
    convert_f32_bf16_kernel<<<(n + 255) / 256, 256, 0, stream>>>(Wo, wo_bf, n);
    int nu = VQ * CQ;
    convert_f32_bf16_kernel<<<(nu + 255) / 256, 256, 0, stream>>>(Wu, wu_bf, nu);
  }
  // 2) h = embed[x] + pos
  embed_kernel<<<(MQ * CQ + 255) / 256, 256, 0, stream>>>(x, embedw, pos, h_f, h_bf);

  dim3 gemmGridC(CQ / 64, MQ / 256);   // N = C = 1024
  dim3 gemmGridV(VQ / 64, MQ / 256);   // N = V = 8192

  for (int l = 0; l < LQ; ++l) {
    const unsigned short* wql = wq_bf + (size_t)l * CQ * CQ;
    const unsigned short* wkl = wk_bf + (size_t)l * CQ * CQ;
    const unsigned short* wvl = wv_bf + (size_t)l * CQ * CQ;
    const unsigned short* wol = wo_bf + (size_t)l * CQ * CQ;
    // 3) projections (bf16 out)
    gemm_bf16_kernel<<<gemmGridC, 256, 0, stream>>>(
        h_bf, wql, bq + l * CQ, nullptr, nullptr, q_bf, MQ, CQ, CQ);
    gemm_bf16_kernel<<<gemmGridC, 256, 0, stream>>>(
        h_bf, wkl, bk + l * CQ, nullptr, nullptr, k_bf, MQ, CQ, CQ);
    gemm_bf16_kernel<<<gemmGridC, 256, 0, stream>>>(
        h_bf, wvl, bv + l * CQ, nullptr, nullptr, v_bf, MQ, CQ, CQ);
    // 4) V transpose for contiguous B-frags in attention
    vtrans_kernel<<<(BQ * HQ * DQ * TQ + 255) / 256, 256, 0, stream>>>(v_bf, vt_bf);
    // 5) flash attention
    attn_kernel<<<(BQ * HQ * (TQ / 16)) / 8, 256, 0, stream>>>(q_bf, k_bf, vt_bf, ctx_bf);
    // 6) output projection + residual -> new h (f32 + bf16)
    gemm_bf16_kernel<<<gemmGridC, 256, 0, stream>>>(
        ctx_bf, wol, bo + l * CQ, h_f, h_f, h_bf, MQ, CQ, CQ);
  }
  // 7) logits = h @ Wu^T  (straight into d_out)
  gemm_bf16_kernel<<<gemmGridV, 256, 0, stream>>>(
      h_bf, wu_bf, nullptr, nullptr, logits, nullptr, MQ, VQ, CQ);
  // 8) loss = mean NLL
  zero_loss_kernel<<<1, 1, 0, stream>>>(lossp);
  loss_kernel<<<MQ, 256, 0, stream>>>(logits, target, lossp);
}
